// GCN_81088982548586
// MI455X (gfx1250) — compile-verified, matrix operand order checked by reading, not run on hardware
//
#include <hip/hip_runtime.h>
#include <hip/hip_bf16.h>

typedef __attribute__((ext_vector_type(16))) _Float16 v16h;
typedef __attribute__((ext_vector_type(8)))  float    v8f;

#define NHID 16
#define INDIM 64
#define EHID 64

// ---------------------------------------------------------------------------
// zero fill (grid-stride)
// ---------------------------------------------------------------------------
__global__ __launch_bounds__(256) void zero_f32(float* __restrict__ p, long long n) {
  long long i = (long long)blockIdx.x * blockDim.x + threadIdx.x;
  long long s = (long long)gridDim.x * blockDim.x;
  for (; i < n; i += s) p[i] = 0.f;
}

// ---------------------------------------------------------------------------
// WL edge net via WMMA f16: per wave, a 16-edge tile.
//   h1 = relu(X1 @ W), h2 = relu(X2 @ W), ew = sigmoid(dot(h1,h2)/8)
// A (16x32 f16, K padded 16->32): lanes 0-15: M=lane, K=0..7 in v0-3;
//                                 lanes16-31: M=lane-16, K=8..15; v4-7 zero.
// B (32x16 f16): same K split with N=lane%16.
// C (16x16 f32): vgpr r: lanes0-15 -> (M=r,N=lane); lanes16-31 -> (M=r+8,N=lane-16)
// Fused: atomicAdd of edge weight into deg[src[e]]; next-tile prefetch.
// ---------------------------------------------------------------------------
__global__ __launch_bounds__(256) void edge_net_wmma(
    const float* __restrict__ ein,    // [E,32]
    const float* __restrict__ wproj,  // [16,64]
    const int*   __restrict__ src,    // [E]
    float* __restrict__ edge_weight,  // [E] (into d_out)
    float* __restrict__ deg,          // [N] (atomic)
    int E)
{
  const int lane  = threadIdx.x & 31;
  const int wav   = threadIdx.x >> 5;
  const int laneM = lane & 15;
  const int hi    = lane >> 4;
  const int c0    = hi * 8;          // K sub-range owned by this lane half

  // B fragments: 4 column blocks of the 16x64 projection, K padded to 32.
  v16h bfr[4];
#pragma unroll
  for (int nb = 0; nb < 4; ++nb) {
    v16h b = {};
#pragma unroll
    for (int i = 0; i < 8; ++i)
      b[i] = (_Float16)wproj[(c0 + i) * EHID + nb * 16 + laneM];
    bfr[nb] = b;
  }

  const int tiles  = (E + 15) >> 4;
  const int wglob  = blockIdx.x * (blockDim.x >> 5) + wav;
  const int nWaves = gridDim.x * (blockDim.x >> 5);
  const v8f  cz    = {};

  for (int tile = wglob; tile < tiles; tile += nWaves) {
    const int base = tile << 4;
    int er = base + laneM;
    if (er >= E) er = E - 1;                 // clamp (tail tile)
    const float* row = ein + (size_t)er * 32;

    // prefetch next tile's row segments (global_prefetch_b8 pulls whole lines)
    int nt = tile + nWaves;
    if (nt < tiles) {
      const float* nrow = ein + (size_t)(((nt << 4) + laneM)) * 32;
      __builtin_prefetch(nrow + c0, 0, 3);
      __builtin_prefetch(nrow + 16 + c0, 0, 3);
    }

    v16h a1 = {}, a2 = {};
#pragma unroll
    for (int i = 0; i < 8; ++i) {
      a1[i] = (_Float16)row[c0 + i];        // first endpoint half
      a2[i] = (_Float16)row[16 + c0 + i];   // second endpoint half
    }

    float p[8] = {0.f,0.f,0.f,0.f,0.f,0.f,0.f,0.f};
#pragma unroll
    for (int nb = 0; nb < 4; ++nb) {
      v8f c1 = __builtin_amdgcn_wmma_f32_16x16x32_f16(
          false, a1, false, bfr[nb], (short)0, cz, false, false);
      v8f c2 = __builtin_amdgcn_wmma_f32_16x16x32_f16(
          false, a2, false, bfr[nb], (short)0, cz, false, false);
#pragma unroll
      for (int r = 0; r < 8; ++r) {
        float h1 = c1[r] > 0.f ? c1[r] : 0.f;
        float h2 = c2[r] > 0.f ? c2[r] : 0.f;
        p[r] += h1 * h2;
      }
    }

    // reduce over the 16 N-columns held by each 16-lane group
#pragma unroll
    for (int m = 1; m < 16; m <<= 1) {
#pragma unroll
      for (int r = 0; r < 8; ++r) p[r] += __shfl_xor(p[r], m, 32);
    }

    // lanes 0..7 own edges 0..7 (rows M=r); lanes 16..23 own edges 8..15 (M=r+8)
    int t = -1;
    if (lane < 8)                     t = lane;
    else if (lane >= 16 && lane < 24) t = lane - 8;
    if (t >= 0) {
      int e = base + t;
      if (e < E) {
        int k = t & 7;
        float s = p[0];
#pragma unroll
        for (int r = 1; r < 8; ++r) if (k == r) s = p[r];
        float w = 1.f / (1.f + __expf(-0.125f * s));   // scale = 1/sqrt(64)
        edge_weight[e] = w;
        atomicAdd(deg + src[e], w);
      }
    }
  }
}

// ---------------------------------------------------------------------------
// deg -> dis = deg>0 ? rsqrt(max(deg,1e-12)) : 0   (in place)
// ---------------------------------------------------------------------------
__global__ __launch_bounds__(256) void deg_to_dis(float* __restrict__ deg, int N) {
  int i = blockIdx.x * blockDim.x + threadIdx.x;
  if (i >= N) return;
  float d = deg[i];
  deg[i] = d > 0.f ? rsqrtf(fmaxf(d, 1e-12f)) : 0.f;
}

// norm[e] = -dis[src]*w[e]*dis[dst]
__global__ __launch_bounds__(256) void edge_norm(
    const float* __restrict__ dis, const float* __restrict__ ew,
    const int* __restrict__ src, const int* __restrict__ dst,
    float* __restrict__ norm, int E)
{
  int e = blockIdx.x * blockDim.x + threadIdx.x;
  if (e >= E) return;
  norm[e] = -dis[src[e]] * ew[e] * dis[dst[e]];
}

// ---------------------------------------------------------------------------
// prop: out[dst][0..F) += norm[e] * t[src*tStride + tOff + 0..F)
// thread = (edge, 4-float chunk).  Source rows may live inside a wider buffer.
// ---------------------------------------------------------------------------
template <int F>
__global__ __launch_bounds__(256) void prop_scatter(
    const float* __restrict__ t, int tStride, int tOff,
    const float* __restrict__ norm,
    const int* __restrict__ src, const int* __restrict__ dst,
    float* __restrict__ out, int E)
{
  constexpr int F4 = F / 4;
  int id = blockIdx.x * blockDim.x + threadIdx.x;
  int e  = id / F4;
  if (e >= E) return;
  int c  = id - e * F4;
  float w = norm[e];
  int s = src[e], d = dst[e];
  const float4 v = *(const float4*)(t + (size_t)s * tStride + tOff + c * 4);
  float* o = out + (size_t)d * F + c * 4;
  atomicAdd(o + 0, w * v.x);
  atomicAdd(o + 1, w * v.y);
  atomicAdd(o + 2, w * v.z);
  atomicAdd(o + 3, w * v.w);
}

// ---------------------------------------------------------------------------
// layer-0 projection: A0 = X@w[0] -> a0 [N,16];  [A1|A2] = X@[w[1],w[2]] -> a12 [N,32]
// (prop commutes with right-multiplication, so project 64->16 BEFORE scattering)
// ---------------------------------------------------------------------------
__global__ __launch_bounds__(256) void cheb0_project(
    const float* __restrict__ x, const float* __restrict__ w,  // [3,64,16]
    float* __restrict__ a0, float* __restrict__ a12, int N)
{
  __shared__ float sw[3 * INDIM * NHID];
  for (int i = threadIdx.x; i < 3 * INDIM * NHID; i += blockDim.x) sw[i] = w[i];
  __syncthreads();
  int n = blockIdx.x * blockDim.x + threadIdx.x;
  if (n >= N) return;
  const float* xr = x + (size_t)n * INDIM;
  float acc0[NHID], acc1[NHID], acc2[NHID];
#pragma unroll
  for (int o = 0; o < NHID; ++o) { acc0[o] = 0.f; acc1[o] = 0.f; acc2[o] = 0.f; }
  for (int k = 0; k < INDIM; ++k) {
    float a = xr[k];
    const float* w0 = sw + k * NHID;
    const float* w1 = sw + INDIM * NHID + k * NHID;
    const float* w2 = sw + 2 * INDIM * NHID + k * NHID;
#pragma unroll
    for (int o = 0; o < NHID; ++o) {
      acc0[o] += a * w0[o];
      acc1[o] += a * w1[o];
      acc2[o] += a * w2[o];
    }
  }
#pragma unroll
  for (int o = 0; o < NHID; ++o) {
    a0[(size_t)n * NHID + o]       = acc0[o];
    a12[(size_t)n * 32 + o]        = acc1[o];
    a12[(size_t)n * 32 + 16 + o]   = acc2[o];
  }
}

// layer-0 final: x = xtemp = relu(A0 + P(A1) + 2*P(P(A2)) - A2)
__global__ __launch_bounds__(256) void cheb0_final(
    const float* __restrict__ a0,   // [N,16]
    const float* __restrict__ a12,  // [N,32]  (A2 in cols 16..31)
    const float* __restrict__ pa,   // [N,32]  (P(A1) cols 0..15)
    const float* __restrict__ ppa2, // [N,16]  (P(P(A2)))
    float* __restrict__ x, float* __restrict__ xtemp, int N)
{
  long long i = (long long)blockIdx.x * blockDim.x + threadIdx.x;
  if (i >= (long long)N * NHID) return;
  long long n = i / NHID;
  int o = (int)(i - n * NHID);
  float v = a0[i] + pa[n * 32 + o] + 2.f * ppa2[i] - a12[n * 32 + 16 + o];
  v = v > 0.f ? v : 0.f;
  x[i] = v;
  xtemp[i] = v;
}

// ---------------------------------------------------------------------------
// middle/final combine: r = relu(xin@w0 + t1@w1 + (2*t2p - xin)@w2), K=16
// mode 0: xtemp_acc += r      mode 1: xout = r
// ---------------------------------------------------------------------------
__global__ __launch_bounds__(256) void cheb_combine16(
    const float* __restrict__ xin, const float* __restrict__ t1,
    const float* __restrict__ t2p, const float* __restrict__ w,  // [3,16,16]
    float* __restrict__ xtemp_acc, float* __restrict__ xout, int N, int mode)
{
  __shared__ float sw[3 * NHID * NHID];
  for (int i = threadIdx.x; i < 3 * NHID * NHID; i += blockDim.x) sw[i] = w[i];
  __syncthreads();
  int n = blockIdx.x * blockDim.x + threadIdx.x;
  if (n >= N) return;
  const float* xr  = xin + (size_t)n * NHID;
  const float* t1r = t1  + (size_t)n * NHID;
  const float* t2r = t2p + (size_t)n * NHID;
  float acc[NHID];
#pragma unroll
  for (int o = 0; o < NHID; ++o) acc[o] = 0.f;
#pragma unroll
  for (int k = 0; k < NHID; ++k) {
    float a0 = xr[k], a1 = t1r[k];
    float a2 = 2.f * t2r[k] - a0;
    const float* w0 = sw + k * NHID;
    const float* w1 = sw + NHID * NHID + k * NHID;
    const float* w2 = sw + 2 * NHID * NHID + k * NHID;
#pragma unroll
    for (int o = 0; o < NHID; ++o)
      acc[o] += a0 * w0[o] + a1 * w1[o] + a2 * w2[o];
  }
#pragma unroll
  for (int o = 0; o < NHID; ++o) {
    float v = acc[o] > 0.f ? acc[o] : 0.f;
    if (mode == 0) xtemp_acc[(size_t)n * NHID + o] += v;
    else           xout[(size_t)n * NHID + o] = v;
  }
}

// ---------------------------------------------------------------------------
// classifier: h = relu(x@W1+b1); h = h*scale+shift (BN eval); out = h@W2+b2
// ---------------------------------------------------------------------------
__global__ __launch_bounds__(256) void classifier_head(
    const float* __restrict__ x, const float* __restrict__ w1,
    const float* __restrict__ b1, const float* __restrict__ gam,
    const float* __restrict__ bet, const float* __restrict__ mean,
    const float* __restrict__ var, const float* __restrict__ w2,
    const float* __restrict__ b2, float* __restrict__ out, int N)
{
  __shared__ float sW1[NHID * 128];
  __shared__ float sScale[128], sShift[128], sB1[128], sW2[256];
  for (int i = threadIdx.x; i < NHID * 128; i += blockDim.x) sW1[i] = w1[i];
  for (int i = threadIdx.x; i < 128; i += blockDim.x) {
    float sc  = gam[i] * rsqrtf(var[i] + 1e-5f);
    sScale[i] = sc;
    sShift[i] = bet[i] - mean[i] * sc;
    sB1[i]    = b1[i];
    sW2[2*i]   = w2[2*i];
    sW2[2*i+1] = w2[2*i+1];
  }
  __syncthreads();
  int n = blockIdx.x * blockDim.x + threadIdx.x;
  if (n >= N) return;
  float xi[NHID];
#pragma unroll
  for (int k = 0; k < NHID; ++k) xi[k] = x[(size_t)n * NHID + k];
  float acc0 = 0.f, acc1 = 0.f;
  for (int j = 0; j < 128; ++j) {
    float h = sB1[j];
#pragma unroll
    for (int k = 0; k < NHID; ++k) h += xi[k] * sW1[k * 128 + j];
    h = h > 0.f ? h : 0.f;
    h = h * sScale[j] + sShift[j];
    acc0 += h * sW2[2*j];
    acc1 += h * sW2[2*j+1];
  }
  out[2*(size_t)n]     = acc0 + b2[0];
  out[2*(size_t)n + 1] = acc1 + b2[1];
}

// ---------------------------------------------------------------------------
extern "C" void kernel_launch(void* const* d_in, const int* in_sizes, int n_in,
                              void* d_out, int out_size, void* d_ws, size_t ws_size,
                              hipStream_t stream) {
  const float* features = (const float*)d_in[0];   // [N,64]
  const int*   eidx     = (const int*)  d_in[1];   // [2,E]
  const float* enet_in  = (const float*)d_in[2];   // [E,32]
  const float* cheb_w0  = (const float*)d_in[3];   // [3,64,16]
  const float* cheb_wh  = (const float*)d_in[4];   // [4,3,16,16]
  const float* edge_w   = (const float*)d_in[5];   // [16,64]
  const float* cls_w1   = (const float*)d_in[6];
  const float* cls_b1   = (const float*)d_in[7];
  const float* bn_gam   = (const float*)d_in[8];
  const float* bn_bet   = (const float*)d_in[9];
  const float* bn_mean  = (const float*)d_in[10];
  const float* bn_var   = (const float*)d_in[11];
  const float* cls_w2   = (const float*)d_in[12];
  const float* cls_b2   = (const float*)d_in[13];

  const int N = in_sizes[0] / INDIM;
  const int E = in_sizes[1] / 2;
  const int* src = eidx;
  const int* dst = eidx + E;

  float* out = (float*)d_out;           // [N,2] logits
  float* ew  = out + (size_t)2 * N;     // [E] edge weights (tuple slot 2)

  // workspace carve-up
  float* deg   = (float*)d_ws;                    // N    (becomes dis in place)
  float* norm  = deg  + N;                        // E
  float* buf0  = norm + E;                        // N*16  A0
  float* bufAB = buf0 + (size_t)N * NHID;         // N*32  [A1|A2]  (mid: t1)
  float* bufC  = bufAB + (size_t)N * 32;          // N*32  P([A1|A2])
  float* bufD  = bufC + (size_t)N * 32;           // N*16  P(P(A2)) (mid: t2)
  float* xbuf  = bufD + (size_t)N * NHID;         // N*16
  float* xtemp = xbuf + (size_t)N * NHID;         // N*16

  const int TB = 256;
  const int nBlkN   = (N + TB - 1) / TB;
  const int nBlkE   = (E + TB - 1) / TB;
  const int nBlkN16 = ((N * NHID) + TB - 1) / TB;
  const int nBlkS16 = ((E * (NHID / 4)) + TB - 1) / TB;
  const int nBlkS32 = ((E * (32 / 4)) + TB - 1) / TB;

  // 1) edge weights + degree
  zero_f32<<<2048, TB, 0, stream>>>(deg, (long long)N);
  edge_net_wmma<<<2048, TB, 0, stream>>>(enet_in, edge_w, src, ew, deg, E);
  deg_to_dis<<<nBlkN, TB, 0, stream>>>(deg, N);
  edge_norm<<<nBlkE, TB, 0, stream>>>(deg, ew, src, dst, norm, E);

  // 2) ChebConv layer 0, reordered: project 64->16 first, then 32/16-wide props
  cheb0_project<<<nBlkN, TB, 0, stream>>>(features, cheb_w0, buf0, bufAB, N);
  zero_f32<<<2048, TB, 0, stream>>>(bufC, (long long)N * 32);
  prop_scatter<32><<<nBlkS32, TB, 0, stream>>>(bufAB, 32, 0, norm, src, dst, bufC, E);
  zero_f32<<<2048, TB, 0, stream>>>(bufD, (long long)N * NHID);
  prop_scatter<NHID><<<nBlkS16, TB, 0, stream>>>(bufC, 32, 16, norm, src, dst, bufD, E);
  cheb0_final<<<nBlkN16, TB, 0, stream>>>(buf0, bufAB, bufC, bufD, xbuf, xtemp, N);

  // 3) middle layers i = 0..2: x = relu(cheb(xtemp, wh[i])); xtemp += x
  for (int i = 0; i < 3; ++i) {
    const float* wh = cheb_wh + (size_t)i * 3 * NHID * NHID;
    zero_f32<<<2048, TB, 0, stream>>>(bufAB, (long long)N * NHID);
    prop_scatter<NHID><<<nBlkS16, TB, 0, stream>>>(xtemp, NHID, 0, norm, src, dst, bufAB, E);
    zero_f32<<<2048, TB, 0, stream>>>(bufD, (long long)N * NHID);
    prop_scatter<NHID><<<nBlkS16, TB, 0, stream>>>(bufAB, NHID, 0, norm, src, dst, bufD, E);
    cheb_combine16<<<nBlkN, TB, 0, stream>>>(xtemp, bufAB, bufD, wh, xtemp, xbuf, N, 0);
  }

  // 4) final layer: x = relu(cheb(xtemp, wh[3]))
  {
    const float* wh = cheb_wh + (size_t)3 * 3 * NHID * NHID;
    zero_f32<<<2048, TB, 0, stream>>>(bufAB, (long long)N * NHID);
    prop_scatter<NHID><<<nBlkS16, TB, 0, stream>>>(xtemp, NHID, 0, norm, src, dst, bufAB, E);
    zero_f32<<<2048, TB, 0, stream>>>(bufD, (long long)N * NHID);
    prop_scatter<NHID><<<nBlkS16, TB, 0, stream>>>(bufAB, NHID, 0, norm, src, dst, bufD, E);
    cheb_combine16<<<nBlkN, TB, 0, stream>>>(xtemp, bufAB, bufD, wh, nullptr, xbuf, N, 1);
  }

  // 5) classifier head
  classifier_head<<<nBlkN, TB, 0, stream>>>(xbuf, cls_w1, cls_b1, bn_gam, bn_bet,
                                            bn_mean, bn_var, cls_w2, cls_b2, out, N);
}
